// TacStModel_46016279609777
// MI455X (gfx1250) — matrix operand order repaired and checked by reading, not compiled, not run on hardware
//
#include <hip/hip_runtime.h>
#include <hip/hip_bf16.h>

// ---------------------------------------------------------------------------
// TacSt two-level GRU for MI455X (gfx1250, wave32, WMMA bf16).
//   Kernel 1: AST GRU. 136 blocks x 256 thr; each block owns 16 sequences,
//             weights bf16-resident in LDS, 128 recurrent steps of WMMA GEMMs.
//   Kernel 2: proj + context GRU + head. 8 blocks x 256 thr, 17 steps.
// ---------------------------------------------------------------------------

typedef __attribute__((ext_vector_type(16))) __bf16 v16bf;
typedef __attribute__((ext_vector_type(8)))  __bf16 v8bf;
typedef __attribute__((ext_vector_type(8)))  float  v8f;

#define B_       128
#define NAST_    17
#define L_       128
#define S_       128
#define OUT_     3
#define TILE_M   16      // sequences per workgroup
#define THREADS  256     // 8 wave32s

// ---- WMMA fragment helpers (layouts per CDNA5 ISA 7.12.2, wave32) ----------

// A: 16x32 bf16 tile, row-major source (ld elements). lanes 0-15: K 0-7,16-23;
// lanes 16-31: K 8-15,24-31 (khalf = lane>>4).
__device__ __forceinline__ v16bf frag_a(const __bf16* A, int ld, int k0, int lane) {
    const __bf16* p = A + (lane & 15) * ld + k0 + (lane >> 4) * 8;
    v8bf lo = *(const v8bf*)p;          // K = k0+khalf*8 .. +7
    v8bf hi = *(const v8bf*)(p + 16);   // K = k0+16+khalf*8 .. +7
    v16bf f;
#pragma unroll
    for (int j = 0; j < 8; ++j) { f[j] = lo[j]; f[j + 8] = hi[j]; }
    return f;
}

// B: 32x16 bf16 tile of W^T where W is [N][K] row-major. lane holds column
// n0+(lane&15); lanes 0-15: K 0-7,16-23; lanes 16-31: K 8-15,24-31.
__device__ __forceinline__ v16bf frag_b(const __bf16* W, int ld, int n0, int k0, int lane) {
    const __bf16* p = W + (n0 + (lane & 15)) * ld + k0 + (lane >> 4) * 8;
    v8bf lo = *(const v8bf*)p;
    v8bf hi = *(const v8bf*)(p + 16);
    v16bf f;
#pragma unroll
    for (int j = 0; j < 8; ++j) { f[j] = lo[j]; f[j + 8] = hi[j]; }
    return f;
}

// D: 16x16 f32. VGPR r, lane l -> element [r + 8*(l>>4)][n0 + (l&15)].
__device__ __forceinline__ void store_d(float* G, int ld, int n0, int lane, v8f acc) {
    int m0 = (lane >> 4) * 8;
    int n  = n0 + (lane & 15);
#pragma unroll
    for (int r = 0; r < 8; ++r) G[(m0 + r) * ld + n] = acc[r];
}

// ---- GRU building blocks ---------------------------------------------------

__device__ __forceinline__ float sig_(float x)  { return 1.0f / (1.0f + __expf(-x)); }
__device__ __forceinline__ float tanh_(float x) { return 1.0f - 2.0f / (__expf(2.0f * x) + 1.0f); }

// gi = A1 @ Wih^T, gh = A2 @ Whh^T.  48 (gate,ntile) tiles over 8 waves.
__device__ __forceinline__ void gemm_gates(const __bf16* A1, const __bf16* A2,
                                           const __bf16* Wih, const __bf16* Whh,
                                           float* GI, float* GH, int wave, int lane) {
#pragma unroll
    for (int i = 0; i < 6; ++i) {
        int  tIdx = wave * 6 + i;
        bool isGH = tIdx >= 24;
        int  n0   = (isGH ? tIdx - 24 : tIdx) * 16;
        const __bf16* A = isGH ? A2 : A1;
        const __bf16* W = isGH ? Whh : Wih;
        v8f acc = {};
#pragma unroll
        for (int k0 = 0; k0 < S_; k0 += 32) {
            v16bf a = frag_a(A, S_, k0, lane);
            v16bf b = frag_b(W, S_, n0, k0, lane);
            acc = __builtin_amdgcn_wmma_f32_16x16x32_bf16(
                false, a, false, b, (short)0, acc, false, false);
        }
        store_d(isGH ? GH : GI, 3 * S_, n0, lane, acc);
    }
}

// GRUCell elementwise update: 2048 elems / 256 threads = 8 each.
__device__ __forceinline__ void gru_update(float* h, __bf16* hbf,
                                           const float* GI, const float* GH,
                                           const float* bih, const float* bhh, int tid) {
#pragma unroll
    for (int e = 0; e < 8; ++e) {
        int flat = tid * 8 + e;
        int row  = flat >> 7, col = flat & 127;
        const float* gi = GI + row * (3 * S_);
        const float* gh = GH + row * (3 * S_);
        float ir = gi[col]          + bih[col];
        float iz = gi[col + S_]     + bih[col + S_];
        float in = gi[col + 2 * S_] + bih[col + 2 * S_];
        float hr = gh[col]          + bhh[col];
        float hz = gh[col + S_]     + bhh[col + S_];
        float hn = gh[col + 2 * S_] + bhh[col + 2 * S_];
        float r = sig_(ir + hr);
        float z = sig_(iz + hz);
        float n = tanh_(in + r * hn);
        float hv   = h[row * S_ + col];
        float hnew = (1.0f - z) * n + z * hv;
        h[row * S_ + col]   = hnew;
        hbf[row * S_ + col] = (__bf16)hnew;
    }
}

__device__ __forceinline__ void load_w_bf16(__bf16* dst, const float* src, int n, int tid) {
    for (int i = tid; i < n; i += THREADS) dst[i] = (__bf16)src[i];
}

// ---- Kernel 1: AST-level GRU ----------------------------------------------
// LDS: Wih 96K + Whh 96K + X 4K + Hbf 4K + H 8K + GI 24K + GH 24K = 256 KB
#define K1_SMEM (2 * 384 * 128 * 2 + 2 * 16 * 128 * 2 + 16 * 128 * 4 + 2 * 16 * 384 * 4)

__global__ __launch_bounds__(THREADS) void ast_gru_kernel(
    const int* __restrict__ token_ids, const float* __restrict__ embed,
    const float* __restrict__ Wih, const float* __restrict__ Whh,
    const float* __restrict__ bih, const float* __restrict__ bhh,
    const float* __restrict__ h0, float* __restrict__ ast_emb) {
    extern __shared__ __align__(16) char smem[];
    __bf16* sWih = (__bf16*)smem;                 // [384][128]
    __bf16* sWhh = sWih + 384 * S_;
    __bf16* sX   = sWhh + 384 * S_;               // [16][128]
    __bf16* sHb  = sX   + TILE_M * S_;
    float*  sH   = (float*)(sHb + TILE_M * S_);   // [16][128]
    float*  sGI  = sH + TILE_M * S_;              // [16][384]
    float*  sGH  = sGI + TILE_M * 3 * S_;

    int tid = threadIdx.x, lane = tid & 31, wave = tid >> 5;
    int seq0 = blockIdx.x * TILE_M;

    load_w_bf16(sWih, Wih, 384 * S_, tid);
    load_w_bf16(sWhh, Whh, 384 * S_, tid);
    for (int i = tid; i < TILE_M * S_; i += THREADS) {
        float v = h0[i & 127];
        sH[i] = v; sHb[i] = (__bf16)v;
    }
    __syncthreads();

    int r = tid >> 4, part = tid & 15;            // 16 thr/row embedding gather
    for (int t = 0; t < L_; ++t) {
        int tok = token_ids[(seq0 + r) * L_ + t];
        const float* erow = embed + tok * S_ + part * 8;
#pragma unroll
        for (int j = 0; j < 8; ++j) sX[r * S_ + part * 8 + j] = (__bf16)erow[j];
        if (t + 1 < L_) {                         // hide next gather's latency
            int tok2 = token_ids[(seq0 + r) * L_ + t + 1];
            __builtin_prefetch(embed + tok2 * S_ + part * 8, 0, 1);
        }
        __syncthreads();
        gemm_gates(sX, sHb, sWih, sWhh, sGI, sGH, wave, lane);
        __syncthreads();
        gru_update(sH, sHb, sGI, sGH, bih, bhh, tid);
        __syncthreads();
    }
    for (int i = tid; i < TILE_M * S_; i += THREADS)
        ast_emb[seq0 * S_ + i] = sH[i];
}

// ---- Kernel 2: proj + context GRU + head ----------------------------------
// LDS: Wih 96K + Whh 96K + PW 32K + A/CXb/Hb 12K + H 8K + CX 8K + GI/GH 48K
//      + flagW 512B  ~= 300.5 KB  (< 320 KB CDNA5 workgroup limit)
#define K2_SMEM (2 * 384 * 128 * 2 + 128 * 128 * 2 + 3 * 16 * 128 * 2 + \
                 2 * 16 * 128 * 4 + 2 * 16 * 384 * 4 + 128 * 4)

__global__ __launch_bounds__(THREADS) void ctx_gru_kernel(
    const float* __restrict__ ast_emb,
    const float* __restrict__ Wih, const float* __restrict__ Whh,
    const float* __restrict__ bih, const float* __restrict__ bhh,
    const float* __restrict__ c0,
    const float* __restrict__ projW, const float* __restrict__ projB,
    const float* __restrict__ finW, const float* __restrict__ finB,
    float* __restrict__ out) {
    extern __shared__ __align__(16) char smem[];
    __bf16* sWih = (__bf16*)smem;                 // [384][128]
    __bf16* sWhh = sWih + 384 * S_;
    __bf16* sPW  = sWhh + 384 * S_;               // [128][128] (proj_W[:, :128])
    __bf16* sA   = sPW  + S_ * S_;                // [16][128] ast emb bf16
    __bf16* sCXb = sA   + TILE_M * S_;            // [16][128] ctx_x bf16
    __bf16* sHb  = sCXb + TILE_M * S_;
    float*  sH   = (float*)(sHb + TILE_M * S_);
    float*  sCX  = sH  + TILE_M * S_;             // [16][128] fp32
    float*  sGI  = sCX + TILE_M * S_;             // [16][384]
    float*  sGH  = sGI + TILE_M * 3 * S_;
    float*  sFW  = sGH + TILE_M * 3 * S_;         // proj_W[:,128] flag column

    int tid = threadIdx.x, lane = tid & 31, wave = tid >> 5;
    int b0 = blockIdx.x * TILE_M;

    load_w_bf16(sWih, Wih, 384 * S_, tid);
    load_w_bf16(sWhh, Whh, 384 * S_, tid);
    for (int i = tid; i < S_ * S_; i += THREADS) {
        int n = i >> 7, k = i & 127;
        sPW[i] = (__bf16)projW[n * (S_ + 1) + k];
    }
    for (int i = tid; i < S_; i += THREADS) sFW[i] = projW[i * (S_ + 1) + S_];
    for (int i = tid; i < TILE_M * S_; i += THREADS) {
        float v = c0[i & 127];
        sH[i] = v; sHb[i] = (__bf16)v;
    }
    __syncthreads();

    for (int t = 0; t < NAST_; ++t) {
        for (int i = tid; i < TILE_M * S_; i += THREADS) {
            int rr = i >> 7, cc = i & 127;
            sA[i] = (__bf16)ast_emb[((b0 + rr) * NAST_ + t) * S_ + cc];
        }
        __syncthreads();
        {   // ctx_x = ast_emb @ proj_W[:, :128]^T  (8 n-tiles, one per wave)
            int n0 = wave * 16;
            v8f acc = {};
#pragma unroll
            for (int k0 = 0; k0 < S_; k0 += 32) {
                v16bf a = frag_a(sA, S_, k0, lane);
                v16bf b = frag_b(sPW, S_, n0, k0, lane);
                acc = __builtin_amdgcn_wmma_f32_16x16x32_bf16(
                    false, a, false, b, (short)0, acc, false, false);
            }
            store_d(sCX, S_, n0, lane, acc);
        }
        __syncthreads();
        for (int i = tid; i < TILE_M * S_; i += THREADS) {
            int cc = i & 127;  // fold concat-flag into effective bias (CONCLU_POS=0)
            float v = sCX[i] + projB[cc] + (t == 0 ? sFW[cc] : 0.0f);
            sCXb[i] = (__bf16)v;
        }
        __syncthreads();
        gemm_gates(sCXb, sHb, sWih, sWhh, sGI, sGH, wave, lane);
        __syncthreads();
        gru_update(sH, sHb, sGI, sGH, bih, bhh, tid);
        __syncthreads();
    }
    if (tid < TILE_M * OUT_) {                    // logits = h @ final_W^T + b
        int rr = tid / OUT_, o = tid % OUT_;
        float acc = finB[o];
        for (int k = 0; k < S_; ++k) acc += finW[o * S_ + k] * sH[rr * S_ + k];
        out[(b0 + rr) * OUT_ + o] = acc;
    }
}

// ---- Host launcher ---------------------------------------------------------

extern "C" void kernel_launch(void* const* d_in, const int* in_sizes, int n_in,
                              void* d_out, int out_size, void* d_ws, size_t ws_size,
                              hipStream_t stream) {
    (void)in_sizes; (void)n_in; (void)out_size; (void)ws_size;
    const int*   token_ids = (const int*)  d_in[0];
    const float* embed     = (const float*)d_in[1];
    const float* ast_Wih   = (const float*)d_in[2];
    const float* ast_Whh   = (const float*)d_in[3];
    const float* ast_bih   = (const float*)d_in[4];
    const float* ast_bhh   = (const float*)d_in[5];
    const float* ast_init  = (const float*)d_in[6];
    const float* ctx_Wih   = (const float*)d_in[7];
    const float* ctx_Whh   = (const float*)d_in[8];
    const float* ctx_bih   = (const float*)d_in[9];
    const float* ctx_bhh   = (const float*)d_in[10];
    const float* ctx_init  = (const float*)d_in[11];
    const float* proj_W    = (const float*)d_in[12];
    const float* proj_b    = (const float*)d_in[13];
    const float* final_W   = (const float*)d_in[14];
    const float* final_b   = (const float*)d_in[15];

    float* ast_emb = (float*)d_ws;               // [B*N_AST][128] fp32, ~1.1 MB

    int nblocks1 = (B_ * NAST_) / TILE_M;        // 136
    ast_gru_kernel<<<nblocks1, THREADS, K1_SMEM, stream>>>(
        token_ids, embed, ast_Wih, ast_Whh, ast_bih, ast_bhh, ast_init, ast_emb);

    int nblocks2 = B_ / TILE_M;                  // 8
    ctx_gru_kernel<<<nblocks2, THREADS, K2_SMEM, stream>>>(
        ast_emb, ctx_Wih, ctx_Whh, ctx_bih, ctx_bhh, ctx_init,
        proj_W, proj_b, final_W, final_b, (float*)d_out);
}
